// TemporalCrossAttentionPrompt_13572096655438
// MI455X (gfx1250) — compile-verified, matrix operand order checked by reading, not run on hardware
//
#include <hip/hip_runtime.h>
#include <math.h>

// ---------------- CDNA5 WMMA types ----------------
typedef __attribute__((ext_vector_type(16))) __bf16 v16bf;
typedef __attribute__((ext_vector_type(8)))  float  v8f;

union FragBF { v16bf v; unsigned int u[8]; };

__device__ __forceinline__ unsigned int f2bf1(float f) {
  unsigned int x = __float_as_uint(f);
  return (x + 0x7fffu + ((x >> 16) & 1u)) >> 16;   // round-to-nearest-even bf16
}
__device__ __forceinline__ unsigned int packbf2(float lo, float hi) {
  return f2bf1(lo) | (f2bf1(hi) << 16);
}

// Problem constants
#define NN   4
#define TT   8
#define LP1  197
#define HH   12
#define DD   64
#define LL   196     // 14*14
#define KPAD 208     // 196 padded to 13*16
#define TPRM 7       // T' = T-1
#define CC   768
#define WN   729     // (2*14-1)^2
#define MTOT 5488    // N*T'*L = 4*7*196

// =====================================================================
// Kernel Z: zero the output
// =====================================================================
__global__ void tca_zero_kernel(float* __restrict__ out, long n) {
  long i = (long)blockIdx.x * blockDim.x + threadIdx.x;
  if (i < n) out[i] = 0.0f;
}

// =====================================================================
// Kernel A: S = scale * Q K^T  (WMMA bf16) -> softmax over keys ->
//           mean over heads -> attn rows (bf16) into workspace.
// grid = (13 q-tiles, 28 n*t, 2 halves), block = 128 (4 waves)
// =====================================================================
__global__ __launch_bounds__(128)
void tca_attn_kernel(const float* __restrict__ qg, const float* __restrict__ kg,
                     unsigned short* __restrict__ attn_ws) {
  __shared__ unsigned int sK[KPAD * 32];   // keys x 32 bf16-pairs over d (26.6 KB)
  __shared__ unsigned int sQ[16 * 32];     // q-tile x 32 pairs           (2 KB)
  __shared__ float        sS[16 * KPAD];   // logits / exp                (13.3 KB)
  __shared__ float        sAcc[16 * KPAD]; // head-mean accumulator       (13.3 KB)
  __shared__ float        sRed[16 * 8];
  __shared__ float        sRowAux[16];

  const int tid   = threadIdx.x;
  const int lane  = tid & 31;
  const int wv    = tid >> 5;
  const int qtile = blockIdx.x;            // 0..12
  const int nt    = blockIdx.y;            // 0..27
  const int hf    = blockIdx.z;            // 0..1
  const int n     = nt / TPRM, tp = nt % TPRM;
  const int t_q   = (hf == 0) ? tp + 1 : tp;
  const int t_k   = (hf == 0) ? tp     : tp + 1;

  for (int i = tid; i < 16 * KPAD; i += 128) sAcc[i] = 0.0f;

  const int row = lane & 15;
  const int pb  = (lane < 16) ? 0 : 4;     // fragment pair base

  for (int h = 0; h < HH; ++h) {
    // ---- stage K (196 x 64 f32 -> bf16 pairs) ----
    for (int i = tid; i < LL * 32; i += 128) {
      int key = i >> 5, pd = i & 31;
      long base = ((long)((n * TT + t_k) * LP1 + 1 + key) * HH + h) * DD;
      float2 kv = ((const float2*)(kg + base))[pd];
      sK[i] = packbf2(kv.x, kv.y);
    }
    for (int i = tid; i < (KPAD - LL) * 32; i += 128) sK[LL * 32 + i] = 0u;
    // ---- stage Q tile (16 x 64) ----
    for (int i = tid; i < 16 * 32; i += 128) {
      int r = i >> 5, pd = i & 31;
      int qi = qtile * 16 + r;
      unsigned int val = 0u;
      if (qi < LL) {
        long base = ((long)((n * TT + t_q) * LP1 + 1 + qi) * HH + h) * DD;
        float2 qv = ((const float2*)(qg + base))[pd];
        val = packbf2(qv.x, qv.y);
      }
      sQ[i] = val;
    }
    // ---- prefetch next head's K slice into cache (global_prefetch_b8) ----
    if (h + 1 < HH) {
      for (int key = tid; key < LL; key += 128) {
        long base = ((long)((n * TT + t_k) * LP1 + 1 + key) * HH + (h + 1)) * DD;
        __builtin_prefetch(kg + base, 0, 1);
      }
    }
    __syncthreads();

    // ---- A fragments (Q tile, contraction chunks d=0..31 / 32..63) ----
    FragBF a0, a1;
#pragma unroll
    for (int v = 0; v < 4; ++v) {
      a0.u[v]     = sQ[row * 32 +      pb + v];
      a0.u[v + 4] = sQ[row * 32 +  8 + pb + v];
      a1.u[v]     = sQ[row * 32 + 16 + pb + v];
      a1.u[v + 4] = sQ[row * 32 + 24 + pb + v];
    }
    // ---- each wave covers k-tiles wv, wv+4, wv+8, (wv+12) ----
    for (int j = wv; j < 13; j += 4) {
      int key = j * 16 + row;
      FragBF b0, b1;
#pragma unroll
      for (int v = 0; v < 4; ++v) {
        b0.u[v]     = sK[key * 32 +      pb + v];
        b0.u[v + 4] = sK[key * 32 +  8 + pb + v];
        b1.u[v]     = sK[key * 32 + 16 + pb + v];
        b1.u[v + 4] = sK[key * 32 + 24 + pb + v];
      }
      v8f acc = {0.f, 0.f, 0.f, 0.f, 0.f, 0.f, 0.f, 0.f};
      acc = __builtin_amdgcn_wmma_f32_16x16x32_bf16(false, a0.v, false, b0.v,
                                                    (short)0, acc, false, false);
      acc = __builtin_amdgcn_wmma_f32_16x16x32_bf16(false, a1.v, false, b1.v,
                                                    (short)0, acc, false, false);
      int mbase = (lane < 16) ? 0 : 8;
      int col = j * 16 + row;
#pragma unroll
      for (int v = 0; v < 8; ++v) sS[(mbase + v) * KPAD + col] = acc[v] * 0.125f;
    }
    __syncthreads();

    // ---- softmax over keys (cols 0..195), accumulate mean over heads ----
    int r = tid >> 3, sub = tid & 7;     // 8 threads per row
    float mx = -__builtin_inff();
    for (int j2 = sub; j2 < LL; j2 += 8) mx = fmaxf(mx, sS[r * KPAD + j2]);
    sRed[r * 8 + sub] = mx;
    __syncthreads();
    if (sub == 0) {
      float mm = sRed[r * 8];
#pragma unroll
      for (int s2 = 1; s2 < 8; ++s2) mm = fmaxf(mm, sRed[r * 8 + s2]);
      sRowAux[r] = mm;
    }
    __syncthreads();
    float rmax = sRowAux[r];
    float psum = 0.0f;
    for (int j2 = sub; j2 < LL; j2 += 8) {
      float e = __expf(sS[r * KPAD + j2] - rmax);
      sS[r * KPAD + j2] = e;
      psum += e;
    }
    sRed[r * 8 + sub] = psum;
    __syncthreads();
    if (sub == 0) {
      float ss = 0.0f;
#pragma unroll
      for (int s2 = 0; s2 < 8; ++s2) ss += sRed[r * 8 + s2];
      sRowAux[r] = ss;
    }
    __syncthreads();
    float inv = 1.0f / (sRowAux[r] * (float)HH);
    for (int j2 = sub; j2 < LL; j2 += 8) sAcc[r * KPAD + j2] += sS[r * KPAD + j2] * inv;
    __syncthreads();
  }

  // ---- store mean-attention rows as bf16 ----
  long rowbase = ((long)hf * 28 + nt) * LL;
  for (int i = tid; i < 16 * LL; i += 128) {
    int r = i / LL, kk = i % LL;
    int qi = qtile * 16 + r;
    if (qi < LL)
      attn_ws[(rowbase + qi) * KPAD + kk] = (unsigned short)f2bf1(sAcc[r * KPAD + kk]);
  }
}

// =====================================================================
// Kernel B: out[n,t,1+q,c] += s @ W  where s[q,widx] is the attention
// gathered through the inverse relative-position map (no index tensor).
// M=5488, K=736 (729 padded), N=768.  grid=(12 n-blk, 86 m-blk), 128 thr.
// =====================================================================
__global__ __launch_bounds__(128)
void tca_outgemm_kernel(const unsigned short* __restrict__ attn, // [MTOT][KPAD] bf16
                        const float* __restrict__ w,             // [729][768]
                        float* __restrict__ out,                 // [4][8][197][768]
                        int tOff) {
  __shared__ unsigned int sA[64 * 16];  // 64 rows  x 16 k-pairs (4 KB)
  __shared__ unsigned int sB[64 * 16];  // 64 cols  x 16 k-pairs (4 KB, transposed W)

  const int tid  = threadIdx.x;
  const int lane = tid & 31, wv = tid >> 5;
  const int m0 = blockIdx.y * 64, c0 = blockIdx.x * 64;
  const int row = lane & 15;
  const int pb  = (lane < 16) ? 0 : 4;

  v8f acc[4];
#pragma unroll
  for (int t = 0; t < 4; ++t) acc[t] = (v8f){0.f,0.f,0.f,0.f,0.f,0.f,0.f,0.f};

  for (int kc = 0; kc < 23; ++kc) {
    const int kbase = kc * 32;
    // ---- stage A: gather attention through inverse IDX map ----
    for (int i = tid; i < 1024; i += 128) {
      int r = i >> 4, pp = i & 15;
      int m = m0 + r;
      unsigned int val = 0u;
      if (m < MTOT) {
        int qq = m % LL;
        int pi = qq / 14, pj = qq % 14;
        unsigned int e0 = 0u, e1 = 0u;
#pragma unroll
        for (int e = 0; e < 2; ++e) {
          int widx = kbase + pp * 2 + e;
          unsigned int vv = 0u;
          if (widx < WN) {
            int ki = pi - (widx / 27) + 13;
            int kj = pj - (widx % 27) + 13;
            if (ki >= 0 && ki < 14 && kj >= 0 && kj < 14)
              vv = attn[(long)m * KPAD + ki * 14 + kj];
          }
          if (e == 0) e0 = vv; else e1 = vv;
        }
        val = e0 | (e1 << 16);
      }
      sA[r * 16 + pp] = val;
    }
    // ---- stage B (transposed, f32->bf16 on the fly) ----
    for (int i = tid; i < 1024; i += 128) {
      int cc = i & 63, pp = i >> 6;
      int col = c0 + cc;
      int w0i = kbase + pp * 2, w1i = w0i + 1;
      float f0 = (w0i < WN) ? w[(long)w0i * CC + col] : 0.0f;
      float f1 = (w1i < WN) ? w[(long)w1i * CC + col] : 0.0f;
      sB[cc * 16 + pp] = packbf2(f0, f1);
    }
    // ---- prefetch next k-step's W rows (global_prefetch_b8) ----
    if (kc + 1 < 23 && tid < 32) {
      int wn = kbase + 32 + tid;
      if (wn < WN) __builtin_prefetch(w + (long)wn * CC + c0, 0, 1);
    }
    __syncthreads();

    // ---- 4 WMMAs per wave: 16 rows x 64 cols ----
    FragBF af;
    int mr = wv * 16 + row;
#pragma unroll
    for (int v = 0; v < 4; ++v) {
      af.u[v]     = sA[mr * 16 +     pb + v];
      af.u[v + 4] = sA[mr * 16 + 8 + pb + v];
    }
#pragma unroll
    for (int ntl = 0; ntl < 4; ++ntl) {
      FragBF bfr;
      int ncol = ntl * 16 + row;
#pragma unroll
      for (int v = 0; v < 4; ++v) {
        bfr.u[v]     = sB[ncol * 16 +     pb + v];
        bfr.u[v + 4] = sB[ncol * 16 + 8 + pb + v];
      }
      acc[ntl] = __builtin_amdgcn_wmma_f32_16x16x32_bf16(false, af.v, false, bfr.v,
                                                         (short)0, acc[ntl], false, false);
    }
    __syncthreads();
  }

  // ---- epilogue: read-modify-write into zeroed output ----
  const int mbase = (lane < 16) ? 0 : 8;
#pragma unroll
  for (int ntl = 0; ntl < 4; ++ntl) {
    int col = c0 + ntl * 16 + row;
#pragma unroll
    for (int v = 0; v < 8; ++v) {
      int m = m0 + wv * 16 + mbase + v;
      if (m < MTOT) {
        int ntIdx = m / LL, qq = m % LL;
        int n = ntIdx / TPRM, tp = ntIdx % TPRM;
        long oidx = ((long)(n * TT + tp + tOff) * LP1 + 1 + qq) * (long)CC + col;
        out[oidx] += acc[ntl][v];
      }
    }
  }
}

// =====================================================================
extern "C" void kernel_launch(void* const* d_in, const int* in_sizes, int n_in,
                              void* d_out, int out_size, void* d_ws, size_t ws_size,
                              hipStream_t stream) {
  const float* q  = (const float*)d_in[0];
  const float* k  = (const float*)d_in[1];
  const float* w1 = (const float*)d_in[2];
  const float* w2 = (const float*)d_in[3];
  float* out = (float*)d_out;
  unsigned short* attn_ws = (unsigned short*)d_ws;   // [2][MTOT][KPAD] bf16 ~ 4.6 MB

  long nOut = (long)out_size;                         // 4*8*197*768
  tca_zero_kernel<<<(int)((nOut + 255) / 256), 256, 0, stream>>>(out, nOut);

  dim3 gA(13, 28, 2);
  tca_attn_kernel<<<gA, 128, 0, stream>>>(q, k, attn_ws);

  dim3 gB(12, 86);   // 768/64 n-blocks, ceil(5488/64) m-blocks
  tca_outgemm_kernel<<<gB, 128, 0, stream>>>(attn_ws, w1, out, 1);
  tca_outgemm_kernel<<<gB, 86 == 0 ? 128 : 128, 0, stream>>>(attn_ws + (long)MTOT * KPAD, w2, out, 0);
}
